// TokenEncoder_4518305595495
// MI455X (gfx1250) — matrix operand order repaired and checked by reading, not compiled, not run on hardware
//
#include <hip/hip_runtime.h>
#include <hip/hip_bf16.h>
#include <math.h>

#define B_   32
#define L_   1024
#define D_   512
#define LP_  (L_ + 2)
#define EPS_ 1e-5f

typedef __attribute__((ext_vector_type(16))) _Float16 v16h;
typedef __attribute__((ext_vector_type(8)))  float    v8f;
typedef __attribute__((ext_vector_type(4)))  float    v4f;

union Frag { v16h h; v4f f[2]; };

__device__ __forceinline__ v8f wmma16(v16h a, v16h b, v8f c) {
  // D = A(16x32 f16) * B(32x16 f16) + C(16x16 f32)
  return __builtin_amdgcn_wmma_f32_16x16x32_f16(false, a, false, b, (short)0, c,
                                                false, false);
}

__device__ __forceinline__ float sig_(float x) { return 1.0f / (1.0f + __expf(-x)); }

// ---------------------------------------------------------------------------
// Weight repack: conv w [3][K=D][N=D] f32 (W-I-O)  ->  wT [3][N][K] f16
// ---------------------------------------------------------------------------
__global__ void pack_conv_w_k(const float* __restrict__ w, _Float16* __restrict__ wt) {
  int i = blockIdx.x * blockDim.x + threadIdx.x;
  if (i >= 3 * D_ * D_) return;
  int n   = i % D_;
  int k   = (i / D_) % D_;
  int tap = i / (D_ * D_);
  wt[((size_t)tap * D_ + n) * D_ + k] = (_Float16)w[i];
}

// LSTM w [2D=1024][4D=2048] f32 -> wT [dir][N=2048][K=1024] f16
__global__ void pack_lstm_w_k(const float* __restrict__ wf, const float* __restrict__ wb,
                              _Float16* __restrict__ wt) {
  const int per = (2 * D_) * (4 * D_);
  int i = blockIdx.x * blockDim.x + threadIdx.x;
  if (i >= 2 * per) return;
  int dir = i / per;
  int j   = i - dir * per;
  int n   = j % (4 * D_);
  int k   = j / (4 * D_);
  const float* w = dir ? wb : wf;
  wt[((size_t)dir * (4 * D_) + n) * (2 * D_) + k] = (_Float16)w[j];
}

// Zero the halo rows (l=0 and l=L+1) of a padded f16 activation buffer
__global__ void zero_pad_rows_k(_Float16* __restrict__ h) {
  int i = blockIdx.x * blockDim.x + threadIdx.x;
  if (i >= 2 * B_ * D_) return;
  int d = i % D_;
  int r = (i / D_) & 1;
  int b = i / (2 * D_);
  int lp = r ? (L_ + 1) : 0;
  h[((size_t)(b * LP_ + lp)) * D_ + d] = (_Float16)0.0f;
}

// Embedding gather -> padded f16 buffer (pads written as zero)
__global__ void embed_pad_k(const int* __restrict__ x, const float* __restrict__ ew,
                            _Float16* __restrict__ hpad) {
  size_t i = (size_t)blockIdx.x * blockDim.x + threadIdx.x;
  if (i >= (size_t)B_ * LP_ * D_) return;
  int dd  = (int)(i % D_);
  size_t rl = i / D_;
  int lp = (int)(rl % LP_);
  int b  = (int)(rl / LP_);
  float v = 0.0f;
  if (lp >= 1 && lp <= L_) {
    int tok = x[b * L_ + lp - 1];
    v = ew[(size_t)tok * D_ + dd];
  }
  hpad[i] = (_Float16)v;
}

// ---------------------------------------------------------------------------
// Conv1D(k=3,SAME) + BN(eval) + ReLU as a software-pipelined WMMA GEMM.
// Block = 256 threads (8 waves). Block owns TWO M-tiles (32 rows of B*L);
// wave w owns N-strip [w*64, w*64+64) as 4 N-tiles -> 8 accumulator tiles.
// K = 3 taps * 512 = 48 WMMA K-steps, fully unrolled, A/B fragments double
// buffered. Each B fragment feeds 2 WMMAs (both M-tiles), so the matrix pipe
// has work in flight while the next 12 b128 loads land (1.5 loads/WMMA).
// ---------------------------------------------------------------------------
__global__ __launch_bounds__(256)
void conv_bn_relu_wmma_k(const _Float16* __restrict__ hin,   // [B][L+2][D] padded f16
                         const _Float16* __restrict__ wT,    // [3][N][K] f16
                         const float* __restrict__ bias,
                         const float* __restrict__ scale,
                         const float* __restrict__ offset,
                         const float* __restrict__ mean,
                         const float* __restrict__ var,
                         _Float16* __restrict__ hout) {      // [B][L+2][D] padded f16
  const int lane = threadIdx.x & 31;
  const int wv   = threadIdx.x >> 5;
  const int m0   = blockIdx.x * 32;          // global row base (B*L rows)
  const int b    = m0 >> 10;                 // / L_  (32 rows stay in one batch)
  const int l0   = m0 & (L_ - 1);
  const int row  = lane & 15;
  const int half = lane >> 4;
  const int n0   = wv * 64;

  // A base for M-tile 0: padded input row (l0+row)+tap; M-tile 1 is +16 rows.
  const _Float16* abase = hin + ((size_t)(b * LP_ + l0 + row)) * D_ + half * 8;
  // weight base for this lane: + tap*D*D + nt*16*D + kb selects the fragment
  const _Float16* wbase = wT + (size_t)(n0 + row) * D_ + half * 8;

  v8f acc[2][4];
#pragma unroll
  for (int mt = 0; mt < 2; ++mt)
#pragma unroll
    for (int i = 0; i < 4; ++i)
#pragma unroll
      for (int j = 0; j < 8; ++j) acc[mt][i][j] = 0.0f;

  Frag A[2][2];     // [buf][mtile]
  Frag Bf[2][4];    // [buf][ntile]

  auto ldstep = [&](int st, Frag* Ar, Frag* Br) {
    const int tap = st >> 4;                  // constant after full unroll
    const int kb  = (st & 15) << 5;
    const _Float16* ap0 = abase + (size_t)tap * D_ + kb;
    const _Float16* ap1 = ap0 + (size_t)16 * D_;
    Ar[0].f[0] = *(const v4f*)ap0;
    Ar[0].f[1] = *(const v4f*)(ap0 + 16);
    Ar[1].f[0] = *(const v4f*)ap1;
    Ar[1].f[1] = *(const v4f*)(ap1 + 16);
    __builtin_prefetch(ap0 + 64, 0, 3);
    const _Float16* wp = wbase + (size_t)tap * D_ * D_ + kb;
#pragma unroll
    for (int nt = 0; nt < 4; ++nt) {
      const v4f* q = (const v4f*)(wp + nt * 16 * D_);
      Br[nt].f[0] = q[0];
      Br[nt].f[1] = q[2];
    }
  };

  ldstep(0, A[0], Bf[0]);
#pragma unroll
  for (int st = 0; st < 48; ++st) {
    const int cur = st & 1;
    if (st < 47) ldstep(st + 1, A[cur ^ 1], Bf[cur ^ 1]);  // overlap with WMMA
#pragma unroll
    for (int nt = 0; nt < 4; ++nt) {
      acc[0][nt] = wmma16(A[cur][0].h, Bf[cur][nt].h, acc[0][nt]);
      acc[1][nt] = wmma16(A[cur][1].h, Bf[cur][nt].h, acc[1][nt]);
    }
  }

  // Fused bias + BN(eval) + ReLU epilogue, write f16 into next padded buffer.
#pragma unroll
  for (int nt = 0; nt < 4; ++nt) {
    int n = n0 + nt * 16 + row;                  // C/D layout: N = lane%16
    float s = scale[n] * rsqrtf(var[n] + EPS_);
    float t = (bias[n] - mean[n]) * s + offset[n];
#pragma unroll
    for (int mt = 0; mt < 2; ++mt) {
#pragma unroll
      for (int r = 0; r < 8; ++r) {
        int lrow = mt * 16 + r + half * 8;       // C/D layout: M = r + 8*(lane/16)
        float y = acc[mt][nt][r] * s + t;
        y = fmaxf(y, 0.0f);
        hout[((size_t)(b * LP_ + l0 + lrow + 1)) * D_ + n] = (_Float16)y;
      }
    }
  }
}

// ---------------------------------------------------------------------------
// Persistent bidirectional LSTM. grid = 2 blocks (dir 0 fwd, dir 1 bwd+reset),
// block = 1024 threads = 32 waves (8 per SIMD -> latency hiding via occupancy,
// deliberately NOT register-double-buffered to avoid spills in the serial
// loop). Wave w owns hidden dims [w*16, w*16+16): it computes i/g/f/o gate
// tiles for those dims (M=32 -> 2 M-tiles, K=1024 -> 32 WMMA K-steps, 8 wmma
// per K-step) then does the cell update locally. LDS holds xh = [x_t | h]
// (32x1024 f16); c lives in registers per lane.
// ---------------------------------------------------------------------------
__global__ __launch_bounds__(1024)
void lstm_wmma_k(const _Float16* __restrict__ xpad,  // [B][L+2][D] padded f16 (conv3 out)
                 const _Float16* __restrict__ wT,    // [2][4D][2D] f16
                 const float* __restrict__ bfwd,
                 const float* __restrict__ bbwd,
                 const int* __restrict__ lengths,
                 float* __restrict__ out) {          // [B][L][2D] f32
  const int dir = blockIdx.x;
  const _Float16* W    = wT + (size_t)dir * (4 * D_) * (2 * D_);
  const float*    bias = dir ? bbwd : bfwd;

  __shared__ _Float16 xh[32][2 * D_];   // 64 KB: cols [0,512)=x_t, [512,1024)=h

  const int tid  = threadIdx.x;
  const int lane = tid & 31;
  const int wv   = tid >> 5;
  const int row  = lane & 15;
  const int half = lane >> 4;
  const int d    = wv * 16 + row;       // this lane's hidden dim (gate column mod D)

  const int sm  = tid >> 5;             // staging: row of xh this thread copies
  const int sc0 = (tid & 31) * 16;      // staging: 16-half chunk base

  // init h part of xh to zero
  {
    uint4 z = {0u, 0u, 0u, 0u};
    uint4* dz = (uint4*)&xh[sm][D_ + sc0];
    dz[0] = z; dz[1] = z;
  }

  float creg[2][8];
#pragma unroll
  for (int mt = 0; mt < 2; ++mt)
#pragma unroll
    for (int r = 0; r < 8; ++r) creg[mt][r] = 0.0f;

  const float bi = bias[d];
  const float bg = bias[D_ + d];
  const float bf = bias[2 * D_ + d];
  const float bo = bias[3 * D_ + d];

  __syncthreads();

  for (int s = 0; s < L_; ++s) {
    const int pos = dir ? (L_ - 1 - s) : s;   // original sequence index

    // ---- stage x_t (and apply bwd reset to h before the cell) ----
    {
      const uint4* src = (const uint4*)(xpad + ((size_t)sm * LP_ + pos + 1) * D_ + sc0);
      uint4* dst = (uint4*)&xh[sm][sc0];
      dst[0] = src[0]; dst[1] = src[1];
      if (dir && (pos >= lengths[sm] - 1)) {   // ResetCore: zero h where mask
        uint4 z = {0u, 0u, 0u, 0u};
        uint4* dz = (uint4*)&xh[sm][D_ + sc0];
        dz[0] = z; dz[1] = z;
      }
    }
    __syncthreads();

    // ---- gate GEMM: [32 x 1024] @ W^T slice -> 8 accum tiles ----
    v8f acc[2][4];
#pragma unroll
    for (int mt = 0; mt < 2; ++mt)
#pragma unroll
      for (int g = 0; g < 4; ++g)
#pragma unroll
        for (int j = 0; j < 8; ++j) acc[mt][g][j] = 0.0f;

    for (int kb = 0; kb < 2 * D_; kb += 32) {
      Frag A0, A1;
      A0.f[0] = *(const v4f*)&xh[row][kb + half * 8];
      A0.f[1] = *(const v4f*)&xh[row][kb + 16 + half * 8];
      A1.f[0] = *(const v4f*)&xh[16 + row][kb + half * 8];
      A1.f[1] = *(const v4f*)&xh[16 + row][kb + 16 + half * 8];
#pragma unroll
      for (int g = 0; g < 4; ++g) {
        const _Float16* wrow = W + ((size_t)g * D_ + d) * (2 * D_) + kb + half * 8;
        Frag Bf;
        Bf.f[0] = ((const v4f*)wrow)[0];
        Bf.f[1] = ((const v4f*)wrow)[2];
        acc[0][g] = wmma16(A0.h, Bf.h, acc[0][g]);
        acc[1][g] = wmma16(A1.h, Bf.h, acc[1][g]);
      }
    }
    __syncthreads();   // all reads of xh done before h is overwritten

    // ---- pointwise cell update (haiku order i,g,f,o; forget bias +1) ----
#pragma unroll
    for (int mt = 0; mt < 2; ++mt) {
#pragma unroll
      for (int r = 0; r < 8; ++r) {
        int m = mt * 16 + r + half * 8;          // batch row (C/D layout)
        bool rst = dir && (pos >= lengths[m] - 1);
        float cold = rst ? 0.0f : creg[mt][r];
        float iv = sig_(acc[mt][0][r] + bi);
        float gv = tanhf(acc[mt][1][r] + bg);
        float fv = sig_(acc[mt][2][r] + bf + 1.0f);
        float ov = sig_(acc[mt][3][r] + bo);
        float cn = fv * cold + iv * gv;
        float hn = ov * tanhf(cn);
        creg[mt][r] = cn;
        xh[m][D_ + d] = (_Float16)hn;            // feed next step
        out[((size_t)m * L_ + pos) * (2 * D_) + (size_t)dir * D_ + d] = hn;
      }
    }
    __syncthreads();
  }
}

// ---------------------------------------------------------------------------
extern "C" void kernel_launch(void* const* d_in, const int* in_sizes, int n_in,
                              void* d_out, int out_size, void* d_ws, size_t ws_size,
                              hipStream_t stream) {
  (void)in_sizes; (void)n_in; (void)out_size; (void)ws_size;

  const int*   x        = (const int*)  d_in[0];
  const int*   lengths  = (const int*)  d_in[1];
  const float* embed_w  = (const float*)d_in[2];
  const float* conv_w[3]  = {(const float*)d_in[3], (const float*)d_in[5], (const float*)d_in[7]};
  const float* conv_b[3]  = {(const float*)d_in[4], (const float*)d_in[6], (const float*)d_in[8]};
  const float* lstm_fw  = (const float*)d_in[9];
  const float* lstm_fb  = (const float*)d_in[10];
  const float* lstm_bw  = (const float*)d_in[11];
  const float* lstm_bb  = (const float*)d_in[12];
  const float* bn_scale[3] = {(const float*)d_in[13], (const float*)d_in[17], (const float*)d_in[21]};
  const float* bn_off[3]   = {(const float*)d_in[14], (const float*)d_in[18], (const float*)d_in[22]};
  const float* bn_mean[3]  = {(const float*)d_in[15], (const float*)d_in[19], (const float*)d_in[23]};
  const float* bn_var[3]   = {(const float*)d_in[16], (const float*)d_in[20], (const float*)d_in[24]};

  char* ws = (char*)d_ws;
  size_t off = 0;
  auto alloc = [&](size_t bytes) -> void* {
    void* p = ws + off;
    off += (bytes + 255) & ~(size_t)255;
    return p;
  };
  _Float16* hA = (_Float16*)alloc((size_t)B_ * LP_ * D_ * 2);           // padded act A
  _Float16* hB = (_Float16*)alloc((size_t)B_ * LP_ * D_ * 2);           // padded act B
  _Float16* cw = (_Float16*)alloc((size_t)3 * 3 * D_ * D_ * 2);         // packed conv W
  _Float16* lw = (_Float16*)alloc((size_t)2 * (4 * D_) * (2 * D_) * 2); // packed lstm W

  // weight repack (every call; deterministic, graph-capture safe)
  for (int i = 0; i < 3; ++i)
    pack_conv_w_k<<<(3 * D_ * D_ + 255) / 256, 256, 0, stream>>>(
        conv_w[i], cw + (size_t)i * 3 * D_ * D_);
  pack_lstm_w_k<<<(2 * (2 * D_) * (4 * D_) + 255) / 256, 256, 0, stream>>>(
      lstm_fw, lstm_bw, lw);

  zero_pad_rows_k<<<(2 * B_ * D_ + 255) / 256, 256, 0, stream>>>(hB);

  embed_pad_k<<<(int)(((size_t)B_ * LP_ * D_ + 255) / 256), 256, 0, stream>>>(
      x, embed_w, hA);

  const int convGrid = (B_ * L_) / 32;  // one 32-row M-block per workgroup
  conv_bn_relu_wmma_k<<<convGrid, 256, 0, stream>>>(
      hA, cw, conv_b[0], bn_scale[0], bn_off[0], bn_mean[0], bn_var[0], hB);
  conv_bn_relu_wmma_k<<<convGrid, 256, 0, stream>>>(
      hB, cw + (size_t)1 * 3 * D_ * D_, conv_b[1], bn_scale[1], bn_off[1], bn_mean[1], bn_var[1], hA);
  conv_bn_relu_wmma_k<<<convGrid, 256, 0, stream>>>(
      hA, cw + (size_t)2 * 3 * D_ * D_, conv_b[2], bn_scale[2], bn_off[2], bn_mean[2], bn_var[2], hB);

  lstm_wmma_k<<<2, 1024, 0, stream>>>(hB, lw, lstm_fb, lstm_bb, lengths, (float*)d_out);
}
